// Attention_59889023976189
// MI455X (gfx1250) — matrix operand order repaired and checked by reading, not compiled, not run on hardware
//
#include <hip/hip_runtime.h>
#include <stdint.h>

// ---------------- types & helpers ----------------
typedef __attribute__((ext_vector_type(16))) __bf16 v16bf;
typedef __attribute__((ext_vector_type(8)))  float  v8f;

union FragU { v16bf v; uint4 q[2]; };

__device__ __forceinline__ uint16_t f2bf(float f) {
    uint32_t u = __float_as_uint(f);
    uint32_t r = u + 0x7FFFu + ((u >> 16) & 1u);   // round-to-nearest-even
    return (uint16_t)(r >> 16);
}
__device__ __forceinline__ float bf2f(uint16_t h) {
    return __uint_as_float(((uint32_t)h) << 16);
}

__device__ __forceinline__ v8f wmma_bf16(v16bf a, v16bf b, v8f c) {
    // (neg_a, A, neg_b, B, c_mod, C, reuse_a, reuse_b)
    return __builtin_amdgcn_wmma_f32_16x16x32_bf16(false, a, false, b, (short)0, c,
                                                   false, false);
}

// A-fragment (16x32 bf16): lane = m (0..15) + 16*half; chunks per ISA 7.12.2
__device__ __forceinline__ v16bf load_a_frag(const uint16_t* rowPtr, int k0, int half) {
    FragU f;
    f.q[0] = *(const uint4*)(rowPtr + k0 + half * 8);
    f.q[1] = *(const uint4*)(rowPtr + k0 + 16 + half * 8);
    return f.v;
}
// B-fragment (32x16 bf16): lane = n (0..15) + 16*half; chunks per ISA 7.12.2 (same K order as A)
__device__ __forceinline__ v16bf load_b_frag(const uint16_t* rowPtr, int k0, int half) {
    FragU f;
    f.q[0] = *(const uint4*)(rowPtr + k0 + half * 8);
    f.q[1] = *(const uint4*)(rowPtr + k0 + 16 + half * 8);
    return f.v;
}

// async 16-byte global -> LDS copy (gfx1250 ASYNCcnt path, GV addressing)
__device__ __forceinline__ void async_ld16(uint32_t ldsAddr, const void* gaddr) {
    asm volatile("global_load_async_to_lds_b128 %0, %1, off"
                 :: "v"(ldsAddr), "v"(gaddr)
                 : "memory");
}

// ---------------- constants ----------------
#define HIDDEN   1024
#define HEADS    8
#define HEAD_DIM 128
#define SEQ      4096
#define BATCH    2
#define M_TOT    (BATCH * SEQ)     // 8192
#define BLK_TOK  2048              // block-causal granularity
#define SCALE    0.08838834764831845f  // 1/sqrt(128)

// ---------------- f32 -> bf16 convert ----------------
__global__ void cvt_f32_bf16(const float* __restrict__ src,
                             uint16_t* __restrict__ dst, int n) {
    int i = blockIdx.x * blockDim.x + threadIdx.x;
    if (i < n) dst[i] = f2bf(src[i]);
}

// ---------------- NT GEMM: C[M,N] = A[M,K] * W[N,K]^T  (bf16 in, f32 acc) ---
// block = 256 threads = 8 waves (4 M-waves x 2 N-waves); wave tile 32x32.
// mode 0: f32 to outF row-major [M_TOT, HIDDEN]
// mode 1: bf16 scatter to [B,H,L,D]      (Q/K)
// mode 2: bf16 scatter to [B,H,D,L]      (V transposed)
__global__ __launch_bounds__(256) void gemm_nt_bf16(
    const uint16_t* __restrict__ A, const uint16_t* __restrict__ W,
    float* __restrict__ outF, uint16_t* __restrict__ outB, int mode) {
    const int lane = threadIdx.x & 31;
    const int wave = threadIdx.x >> 5;
    const int ln   = lane & 15;
    const int half = lane >> 4;
    const int wm   = wave & 3;
    const int wn   = wave >> 2;
    const int m0 = blockIdx.y * 128 + wm * 32;
    const int n0 = blockIdx.x * 64  + wn * 32;

    const uint16_t* arow0 = A + (size_t)(m0 + ln)      * HIDDEN;
    const uint16_t* arow1 = A + (size_t)(m0 + 16 + ln) * HIDDEN;
    const uint16_t* wrow0 = W + (size_t)(n0 + ln)      * HIDDEN;
    const uint16_t* wrow1 = W + (size_t)(n0 + 16 + ln) * HIDDEN;

    v8f acc00 = {}, acc01 = {}, acc10 = {}, acc11 = {};
    for (int k0 = 0; k0 < HIDDEN; k0 += 32) {
        if (k0 + 64 < HIDDEN) {   // stream-ahead hint -> global_prefetch_b8
            __builtin_prefetch(arow0 + k0 + 64, 0, 3);
            __builtin_prefetch(wrow0 + k0 + 64, 0, 3);
        }
        v16bf a0 = load_a_frag(arow0, k0, half);
        v16bf a1 = load_a_frag(arow1, k0, half);
        v16bf b0 = load_b_frag(wrow0, k0, half);
        v16bf b1 = load_b_frag(wrow1, k0, half);
        acc00 = wmma_bf16(a0, b0, acc00);
        acc01 = wmma_bf16(a0, b1, acc01);
        acc10 = wmma_bf16(a1, b0, acc10);
        acc11 = wmma_bf16(a1, b1, acc11);
    }

    v8f accs[2][2] = {{acc00, acc01}, {acc10, acc11}};
#pragma unroll
    for (int im = 0; im < 2; ++im)
#pragma unroll
        for (int in = 0; in < 2; ++in)
#pragma unroll
            for (int r = 0; r < 8; ++r) {
                const int row = m0 + im * 16 + r + 8 * half;   // token index
                const int col = n0 + in * 16 + ln;             // feature index
                const float v = accs[im][in][r];
                if (mode == 0) {
                    outF[(size_t)row * HIDDEN + col] = v;
                } else {
                    const int b = row >> 12, l = row & (SEQ - 1);
                    const int h = col >> 7, d = col & (HEAD_DIM - 1);
                    if (mode == 1)
                        outB[(((size_t)(b * HEADS + h)) * SEQ + l) * HEAD_DIM + d] = f2bf(v);
                    else
                        outB[(((size_t)(b * HEADS + h)) * HEAD_DIM + d) * SEQ + l] = f2bf(v);
                }
            }
}

// ---------------- RoPE in-place on Q and K  [B,H,L,D] bf16 ----------------
__global__ void rope_kernel(uint16_t* __restrict__ Q, uint16_t* __restrict__ K,
                            const int* __restrict__ pos) {
    const int total = BATCH * HEADS * SEQ * (HEAD_DIM / 2);
    int i = blockIdx.x * blockDim.x + threadIdx.x;
    if (i >= total) return;
    const int d2 = i & 63;  int t = i >> 6;
    const int l  = t & (SEQ - 1);  t >>= 12;
    const int h  = t & (HEADS - 1);
    const int b  = t >> 3;
    const float p   = (float)pos[b * SEQ + l];
    // inv_freq = 10000^(-2*d2/128)
    const float ang = p * __expf(-(float)(2 * d2) * (9.210340371976184f / 128.0f));
    float sn, cs;
    __sincosf(ang, &sn, &cs);
    const size_t off = (((size_t)(b * HEADS + h)) * SEQ + l) * HEAD_DIM + 2 * d2;
    float x0 = bf2f(Q[off]), x1 = bf2f(Q[off + 1]);
    Q[off]     = f2bf(x0 * cs - x1 * sn);
    Q[off + 1] = f2bf(x0 * sn + x1 * cs);
    x0 = bf2f(K[off]); x1 = bf2f(K[off + 1]);
    K[off]     = f2bf(x0 * cs - x1 * sn);
    K[off + 1] = f2bf(x0 * sn + x1 * cs);
}

// ---------------- flash attention (cooperative LDS staging) ----------------
// 8 waves/block own 8 consecutive 16-row Q tiles of the SAME (b,h). A block's
// 128 q-rows never straddle a 2048-token causal block, so every wave runs the
// same key-tile count -> __syncthreads is safe. K/V tiles are staged once per
// block into LDS via global_load_async_to_lds_b128 (ASYNCcnt) and shared.
__global__ __launch_bounds__(256) void flash_attn(
    const uint16_t* __restrict__ Q, const uint16_t* __restrict__ K,
    const uint16_t* __restrict__ Vt, const unsigned char* __restrict__ amask,
    const int* __restrict__ seg, uint16_t* __restrict__ attnOut) {
    __shared__ uint16_t Ktile[32 * HEAD_DIM];    // 8 KB  [key][d]
    __shared__ uint16_t Vtile[HEAD_DIM * 32];    // 8 KB  [d][key]
    __shared__ uint16_t pbuf[8][16 * 32];        // 8 KB  per-wave P staging

    const int tid  = threadIdx.x;
    const int lane = tid & 31;
    const int wave = tid >> 5;
    const int ln   = lane & 15;
    const int half = lane >> 4;

    const int w  = blockIdx.x * 8 + wave;     // global q-tile id
    const int qt = w & 255;
    const int h  = (w >> 8) & (HEADS - 1);
    const int b  = w >> 11;
    const int qbase = qt * 16;

    const uint16_t* Qh = Q  + ((size_t)(b * HEADS + h)) * SEQ * HEAD_DIM;
    const uint16_t* Kh = K  + ((size_t)(b * HEADS + h)) * SEQ * HEAD_DIM;
    const uint16_t* Vh = Vt + ((size_t)(b * HEADS + h)) * HEAD_DIM * SEQ;

    const uint32_t ldsK = (uint32_t)(uintptr_t)(void*)Ktile;
    const uint32_t ldsV = (uint32_t)(uintptr_t)(void*)Vtile;

    // Q A-fragments for all 4 K-chunks of HEAD_DIM
    v16bf qa[4];
    const uint16_t* qrow = Qh + (size_t)(qbase + ln) * HEAD_DIM;
#pragma unroll
    for (int c = 0; c < 4; ++c) qa[c] = load_a_frag(qrow, 32 * c, half);

    // per-lane row state for rows r + 8*half
    float mrow[8], lrow[8];
    int   segq[8];
#pragma unroll
    for (int r = 0; r < 8; ++r) {
        mrow[r] = -3.0e38f;
        lrow[r] = 0.0f;
        segq[r] = seg[b * SEQ + qbase + r + 8 * half];
    }
    v8f o[8] = {};

    // uniform across the block: blockIdx's 8 q-tiles share one causal block
    const int qblk = qbase >> 11;                   // /2048
    const int nkt  = (qblk + 1) * (BLK_TOK / 32);   // tiles with kblk <= qblk

    uint16_t* pb = pbuf[wave];

    for (int kt = 0; kt < nkt; ++kt) {
        const int kj = kt * 32;

        // ---- cooperative async staging: 512 x 16B for K, same for V ----
#pragma unroll
        for (int rep = 0; rep < 2; ++rep) {
            const int j = tid + rep * 256;
            const int krowI = j >> 4, kchunk = j & 15;       // K: [32][128]
            async_ld16(ldsK + (uint32_t)j * 16,
                       Kh + (size_t)(kj + krowI) * HEAD_DIM + kchunk * 8);
            const int vrow = j >> 2, vc = j & 3;             // V: [128][32]
            async_ld16(ldsV + (uint32_t)j * 16,
                       Vh + (size_t)vrow * SEQ + kj + vc * 8);
        }
        asm volatile("s_wait_asynccnt 0" ::: "memory");
        __syncthreads();

        // ---- S = Q * K^T  (16x32 -> two accumulators), K from LDS ----
        v8f s0 = {}, s1 = {};
        const uint16_t* krow0 = Ktile + (size_t)ln * HEAD_DIM;
        const uint16_t* krow1 = Ktile + (size_t)(16 + ln) * HEAD_DIM;
#pragma unroll
        for (int c = 0; c < 4; ++c) {
            v16bf kb0 = load_b_frag(krow0, 32 * c, half);
            v16bf kb1 = load_b_frag(krow1, 32 * c, half);
            s0 = wmma_bf16(qa[c], kb0, s0);
            s1 = wmma_bf16(qa[c], kb1, s1);
        }

        // masks for this lane's two key columns
        const int k0c = kj + ln, k1c = kj + 16 + ln;
        const bool am0 = amask[b * SEQ + k0c] != 0;
        const bool am1 = amask[b * SEQ + k1c] != 0;
        const int  sk0 = seg[b * SEQ + k0c];
        const int  sk1 = seg[b * SEQ + k1c];

        float p0[8], p1[8], alpha[8];
#pragma unroll
        for (int r = 0; r < 8; ++r) {
            float v0 = s0[r] * SCALE;
            float v1 = s1[r] * SCALE;
            if (!(am0 && sk0 == segq[r])) v0 = -1.0e30f;
            if (!(am1 && sk1 == segq[r])) v1 = -1.0e30f;
            // row max across the 16-lane half (xor masks stay inside the half)
            float mv = fmaxf(v0, v1);
            mv = fmaxf(mv, __shfl_xor(mv, 1));
            mv = fmaxf(mv, __shfl_xor(mv, 2));
            mv = fmaxf(mv, __shfl_xor(mv, 4));
            mv = fmaxf(mv, __shfl_xor(mv, 8));
            const float mnew = fmaxf(mrow[r], mv);
            const float e0 = __expf(v0 - mnew);
            const float e1 = __expf(v1 - mnew);
            float rs = e0 + e1;
            rs += __shfl_xor(rs, 1);
            rs += __shfl_xor(rs, 2);
            rs += __shfl_xor(rs, 4);
            rs += __shfl_xor(rs, 8);
            const float a = __expf(mrow[r] - mnew);
            lrow[r] = lrow[r] * a + rs;
            mrow[r] = mnew;
            alpha[r] = a;
            p0[r] = e0;
            p1[r] = e1;
        }

        // rescale running O
#pragma unroll
        for (int dt = 0; dt < 8; ++dt)
#pragma unroll
            for (int r = 0; r < 8; ++r) o[dt][r] *= alpha[r];

        // stage P (16x32 bf16) row-major into LDS, reload as A-fragment
#pragma unroll
        for (int r = 0; r < 8; ++r) {
            const int row = r + 8 * half;
            pb[row * 32 + ln]      = f2bf(p0[r]);
            pb[row * 32 + 16 + ln] = f2bf(p1[r]);
        }
        asm volatile("s_wait_dscnt 0" ::: "memory");
        FragU pf;
        pf.q[0] = *(const uint4*)(pb + ln * 32 + half * 8);
        pf.q[1] = *(const uint4*)(pb + ln * 32 + 16 + half * 8);

        // ---- O += P * V, V from LDS (rows of Vtile contiguous in keys) ----
#pragma unroll
        for (int dt = 0; dt < 8; ++dt) {
            v16bf vb = load_b_frag(Vtile + (size_t)(dt * 16 + ln) * 32, 0, half);
            o[dt] = wmma_bf16(pf.v, vb, o[dt]);
        }
        __syncthreads();   // protect K/V tiles before next stage (WAR)
    }

    // epilogue: normalize rows and emit bf16 [B, L, HIDDEN]
#pragma unroll
    for (int dt = 0; dt < 8; ++dt)
#pragma unroll
        for (int r = 0; r < 8; ++r) {
            const int qrowi = qbase + r + 8 * half;
            const int col   = h * HEAD_DIM + dt * 16 + ln;
            const float inv = (lrow[r] > 0.0f) ? (1.0f / lrow[r]) : 0.0f;
            attnOut[((size_t)(b * SEQ + qrowi)) * HIDDEN + col] = f2bf(o[dt][r] * inv);
        }
}

// ---------------- driver ----------------
extern "C" void kernel_launch(void* const* d_in, const int* in_sizes, int n_in,
                              void* d_out, int out_size, void* d_ws, size_t ws_size,
                              hipStream_t stream) {
    const float* x  = (const float*)d_in[0];
    const float* wq = (const float*)d_in[1];
    const float* wk = (const float*)d_in[2];
    const float* wv = (const float*)d_in[3];
    const float* wo = (const float*)d_in[4];
    const unsigned char* amask = (const unsigned char*)d_in[5];
    const int* seg = (const int*)d_in[6];
    const int* pos = (const int*)d_in[7];
    float* out = (float*)d_out;

    // workspace carve-up
    char* ws = (char*)d_ws;
    uint16_t* Xb  = (uint16_t*)ws;  ws += (size_t)M_TOT * HIDDEN * 2;      // 16 MB
    uint16_t* Wqb = (uint16_t*)ws;  ws += (size_t)HIDDEN * HIDDEN * 2;     // 2 MB
    uint16_t* Wkb = (uint16_t*)ws;  ws += (size_t)HIDDEN * HIDDEN * 2;
    uint16_t* Wvb = (uint16_t*)ws;  ws += (size_t)HIDDEN * HIDDEN * 2;
    uint16_t* Wob = (uint16_t*)ws;  ws += (size_t)HIDDEN * HIDDEN * 2;
    uint16_t* Qb  = (uint16_t*)ws;  ws += (size_t)M_TOT * HIDDEN * 2;      // [B,H,L,D]
    uint16_t* Kb  = (uint16_t*)ws;  ws += (size_t)M_TOT * HIDDEN * 2;
    uint16_t* Vtb = (uint16_t*)ws;  ws += (size_t)M_TOT * HIDDEN * 2;      // [B,H,D,L]
    uint16_t* Ab  = (uint16_t*)ws;  ws += (size_t)M_TOT * HIDDEN * 2;      // attn bf16

    const int nX = M_TOT * HIDDEN;        // 8,388,608
    const int nW = HIDDEN * HIDDEN;       // 1,048,576
    cvt_f32_bf16<<<(nX + 255) / 256, 256, 0, stream>>>(x, Xb, nX);
    cvt_f32_bf16<<<(nW + 255) / 256, 256, 0, stream>>>(wq, Wqb, nW);
    cvt_f32_bf16<<<(nW + 255) / 256, 256, 0, stream>>>(wk, Wkb, nW);
    cvt_f32_bf16<<<(nW + 255) / 256, 256, 0, stream>>>(wv, Wvb, nW);
    cvt_f32_bf16<<<(nW + 255) / 256, 256, 0, stream>>>(wo, Wob, nW);

    dim3 ggrid(HIDDEN / 64, M_TOT / 128);   // (16, 64)
    gemm_nt_bf16<<<ggrid, 256, 0, stream>>>(Xb, Wqb, nullptr, Qb, 1);
    gemm_nt_bf16<<<ggrid, 256, 0, stream>>>(Xb, Wkb, nullptr, Kb, 1);
    gemm_nt_bf16<<<ggrid, 256, 0, stream>>>(Xb, Wvb, nullptr, Vtb, 2);

    const int nR = BATCH * HEADS * SEQ * (HEAD_DIM / 2);   // 4,194,304
    rope_kernel<<<(nR + 255) / 256, 256, 0, stream>>>(Qb, Kb, pos);

    flash_attn<<<BATCH * HEADS * (SEQ / 16) / 8, 256, 0, stream>>>(
        Qb, Kb, Vtb, amask, seg, Ab);

    gemm_nt_bf16<<<ggrid, 256, 0, stream>>>(Ab, Wob, out, nullptr, 0);
}
